// ComplexMultiHeadAttention_21079699488880
// MI455X (gfx1250) — compile-verified, matrix operand order checked by reading, not run on hardware
//
#include <hip/hip_runtime.h>
#include <hip/hip_bf16.h>

// ---------------------------------------------------------------------------
// ComplexMultiHeadAttention for MI455X (gfx1250), wave32 + WMMA f16->f32,
// async global->LDS (ASYNCcnt) double-buffered staging.
// B=4, S=1024, E=512, H=8, HD=64.
// ---------------------------------------------------------------------------

typedef __attribute__((ext_vector_type(16))) _Float16 v16h;
typedef __attribute__((ext_vector_type(2)))  _Float16 h2;
typedef __attribute__((ext_vector_type(8)))  float    v8f;

#define WMMA_F16(A, B, C) \
  __builtin_amdgcn_wmma_f32_16x16x32_f16(false, (A), false, (B), (short)0, (C), false, false)

constexpr int CB  = 4;
constexpr int CS  = 1024;
constexpr int CE  = 512;
constexpr int CH  = 8;
constexpr int CHD = 64;
constexpr int CM  = CB * CS;          // 4096 rows in the projection GEMMs
#define ATT_SCALE 0.125f              // HD^-0.5 = 1/8

// ---------------------------------------------------------------------------
// CDNA5 async global->LDS copy (bypasses VGPRs, tracked by ASYNCcnt).
// GV mode: 64-bit global address in a VGPR pair, saddr = off.
// ---------------------------------------------------------------------------
__device__ __forceinline__ void async_b128(_Float16* lds, const _Float16* g) {
  unsigned int       l = (unsigned int)(size_t)lds;
  unsigned long long a = (unsigned long long)(size_t)g;
  asm volatile("global_load_async_to_lds_b128 %0, %1, off"
               :: "v"(l), "v"(a) : "memory");
}
__device__ __forceinline__ void wait_async0() {
  asm volatile("s_wait_asynccnt 0" ::: "memory");
}

// ---------------------------------------------------------------------------
// Fragment loaders (per CDNA5 ISA 16-bit A 16x32 layout; B symmetric).
// A frag: lane -> row (M), 8 K-pairs per lane.  K0(v) = 2v + (lane>=16?8:0)
//                                                        + (v>=4?8:0)
// ---------------------------------------------------------------------------
__device__ __forceinline__ v16h frag_rm(const _Float16* base, int ld) {
  // source stored [row][k] with k contiguous (row-major): A tiles, W tiles,
  // K^T tiles, P tiles.
  const int lane = threadIdx.x & 31;
  const int row  = lane & 15;
  const int koff = (lane & 16) ? 8 : 0;
  v16h f;
#pragma unroll
  for (int v = 0; v < 8; ++v) {
    const int k = 2 * v + koff + ((v & 4) ? 8 : 0);
    h2 p = *(const h2*)(base + (size_t)row * ld + k);
    f[2 * v]     = p[0];
    f[2 * v + 1] = p[1];
  }
  return f;
}

__device__ __forceinline__ v16h frag_cm(const _Float16* base, int ld) {
  // source stored [k][col] with col contiguous (V tiles: B[k=key][n=dim]).
  const int lane = threadIdx.x & 31;
  const int col  = lane & 15;
  const int koff = (lane & 16) ? 8 : 0;
  v16h f;
#pragma unroll
  for (int v = 0; v < 8; ++v) {
    const int k = 2 * v + koff + ((v & 4) ? 8 : 0);
    f[2 * v]     = base[(size_t)k * ld + col];
    f[2 * v + 1] = base[(size_t)(k + 1) * ld + col];
  }
  return f;
}

// ---------------------------------------------------------------------------
// fp32 -> f16 conversion
// ---------------------------------------------------------------------------
__global__ void cvt_f32_to_f16(const float* __restrict__ src,
                               _Float16* __restrict__ dst, int n) {
  int i = blockIdx.x * blockDim.x + threadIdx.x;
  if (i < n) dst[i] = (_Float16)src[i];
}

// ---------------------------------------------------------------------------
// Fused complex linear:  Yr = Xr Wr^T - Xi Wi^T + (br - bi)
//                        Yi = Xi Wr^T + Xr Wi^T + (br + bi)
// M=4096, N=512, K=512.  Workgroup tile 128x64, 8 waves, wave tile 32x32.
// Double-buffered async LDS staging: 48 KB LDS.
// ---------------------------------------------------------------------------
__global__ __launch_bounds__(256)
void cplx_gemm_kernel(const _Float16* __restrict__ Ar, const _Float16* __restrict__ Ai,
                      const _Float16* __restrict__ Wr, const _Float16* __restrict__ Wi,
                      const float* __restrict__ Br,   const float* __restrict__ Bi,
                      _Float16* __restrict__ oR16,    _Float16* __restrict__ oI16,
                      float* __restrict__ oRf,        float* __restrict__ oIf) {
  __shared__ __align__(16) _Float16 sAr[2][128 * 32];
  __shared__ __align__(16) _Float16 sAi[2][128 * 32];
  __shared__ __align__(16) _Float16 sWr[2][64 * 32];
  __shared__ __align__(16) _Float16 sWi[2][64 * 32];

  const int m0 = blockIdx.x * 128;
  const int n0 = blockIdx.y * 64;
  const int t  = threadIdx.x;
  const int w  = t >> 5;
  const int mw = (w >> 1) * 32;   // wave M offset: 0,32,64,96
  const int nw = (w & 1) * 32;    // wave N offset: 0,32

  v8f accRa[2][2], accRb[2][2], accI[2][2];
  {
    v8f z = {};
#pragma unroll
    for (int mi = 0; mi < 2; ++mi)
#pragma unroll
      for (int ni = 0; ni < 2; ++ni) {
        accRa[mi][ni] = z; accRb[mi][ni] = z; accI[mi][ni] = z;
      }
  }

  // Async-stage one k-slice (A: 128x32 r/i, W: 64x32 r/i) into buffer `buf`.
  auto stage = [&](int buf, int k0) {
#pragma unroll
    for (int cc = 0; cc < 2; ++cc) {       // A tiles: 512 b128 chunks, 2/thread
      int c   = t + cc * 256;
      int row = c >> 2;
      int col = (c & 3) * 8;
      size_t g = (size_t)(m0 + row) * CE + k0 + col;
      async_b128(&sAr[buf][row * 32 + col], Ar + g);
      async_b128(&sAi[buf][row * 32 + col], Ai + g);
    }
    {                                       // W tiles: 256 b128 chunks, 1/thread
      int row = t >> 2;
      int col = (t & 3) * 8;
      size_t g = (size_t)(n0 + row) * CE + k0 + col;
      async_b128(&sWr[buf][row * 32 + col], Wr + g);
      async_b128(&sWi[buf][row * 32 + col], Wi + g);
    }
  };

  stage(0, 0);
  for (int k0 = 0; k0 < CE; k0 += 32) {
    const int cur = (k0 >> 5) & 1;
    wait_async0();        // my async copies (incl. buffer `cur`) have landed
    __syncthreads();      // everyone's copies landed; prev reads of cur^1 done
    if (k0 + 32 < CE) stage(cur ^ 1, k0 + 32);   // prefetch next slice

    v16h aR[2], aI[2], bR[2], bI[2];
#pragma unroll
    for (int i = 0; i < 2; ++i) {
      aR[i] = frag_rm(&sAr[cur][(mw + i * 16) * 32], 32);
      aI[i] = frag_rm(&sAi[cur][(mw + i * 16) * 32], 32);
      bR[i] = frag_rm(&sWr[cur][(nw + i * 16) * 32], 32);
      bI[i] = frag_rm(&sWi[cur][(nw + i * 16) * 32], 32);
    }
#pragma unroll
    for (int mi = 0; mi < 2; ++mi)
#pragma unroll
      for (int ni = 0; ni < 2; ++ni) {
        accRa[mi][ni] = WMMA_F16(aR[mi], bR[ni], accRa[mi][ni]);
        accRb[mi][ni] = WMMA_F16(aI[mi], bI[ni], accRb[mi][ni]);
        accI[mi][ni]  = WMMA_F16(aI[mi], bR[ni], accI[mi][ni]);
        accI[mi][ni]  = WMMA_F16(aR[mi], bI[ni], accI[mi][ni]);
      }
  }

  // Epilogue: D layout = VGPR r -> M = r + (lane>=16 ? 8 : 0), N = lane&15.
  const int lane = t & 31;
  const int nl   = lane & 15;
  const int mb   = (lane & 16) ? 8 : 0;
#pragma unroll
  for (int mi = 0; mi < 2; ++mi)
#pragma unroll
    for (int ni = 0; ni < 2; ++ni)
#pragma unroll
      for (int r = 0; r < 8; ++r) {
        int m = m0 + mw + mi * 16 + r + mb;
        int n = n0 + nw + ni * 16 + nl;
        float vbr = Br[n], vbi = Bi[n];
        float yr = accRa[mi][ni][r] - accRb[mi][ni][r] + (vbr - vbi);
        float yi = accI[mi][ni][r] + (vbr + vbi);
        size_t off = (size_t)m * CE + n;
        if (oR16) { oR16[off] = (_Float16)yr; oI16[off] = (_Float16)yi; }
        if (oRf)  { oRf[off] = yr; oIf[off] = yi; }
      }
}

// ---------------------------------------------------------------------------
// Flash-style complex attention.  Grid (B*H, S/64), block 128 (4 waves).
// Each wave owns 16 query rows; online softmax over |Q conj(K)^T| * SCALE.
// K/V tiles double-buffered via async global->LDS: 36 KB LDS.
// ---------------------------------------------------------------------------
__global__ __launch_bounds__(128)
void cplx_attn_kernel(const _Float16* __restrict__ Qr, const _Float16* __restrict__ Qi,
                      const _Float16* __restrict__ Kr, const _Float16* __restrict__ Ki,
                      const _Float16* __restrict__ Vr, const _Float16* __restrict__ Vi,
                      _Float16* __restrict__ Or,       _Float16* __restrict__ Oi) {
  __shared__ __align__(16) _Float16 sKr[2][32 * 64];
  __shared__ __align__(16) _Float16 sKi[2][32 * 64];
  __shared__ __align__(16) _Float16 sVr[2][32 * 64];
  __shared__ __align__(16) _Float16 sVi[2][32 * 64];
  __shared__ __align__(16) _Float16 sP[4][16 * 32];

  const int bh = blockIdx.x;
  const int b  = bh / CH;
  const int h  = bh % CH;
  const int q0 = blockIdx.y * 64;
  const int t  = threadIdx.x;
  const int w  = t >> 5;
  const int lane = t & 31;
  const int nl   = lane & 15;
  const int mb   = (lane & 16) ? 8 : 0;
  const size_t hoff = (size_t)h * CHD;

  // Persistent Q fragments (16 rows x 64 dims = 2 K-chunks), straight from HBM.
  v16h qfr[2], qfi[2];
  {
    const _Float16* qr_base = Qr + (size_t)(b * CS + q0 + w * 16) * CE + hoff;
    const _Float16* qi_base = Qi + (size_t)(b * CS + q0 + w * 16) * CE + hoff;
#pragma unroll
    for (int c = 0; c < 2; ++c) {
      qfr[c] = frag_rm(qr_base + c * 32, CE);
      qfi[c] = frag_rm(qi_base + c * 32, CE);
    }
  }

  float rmax[8], rsum[8];
  v8f accOr[4], accOi[4];
  {
    v8f z = {};
#pragma unroll
    for (int i = 0; i < 4; ++i) { accOr[i] = z; accOi[i] = z; }
#pragma unroll
    for (int r = 0; r < 8; ++r) { rmax[r] = -3.0e38f; rsum[r] = 0.0f; }
  }

  // Async-stage K/V (r,i) 32x64 tiles for key chunk `kc` into buffer `buf`.
  auto stageKV = [&](int buf, int kc) {
#pragma unroll
    for (int cc = 0; cc < 2; ++cc) {       // 256 b128 chunks/matrix, 2/thread
      int c   = t + cc * 128;
      int row = c >> 3;
      int col = (c & 7) * 8;
      size_t g = (size_t)(b * CS + kc + row) * CE + hoff + col;
      async_b128(&sKr[buf][row * 64 + col], Kr + g);
      async_b128(&sKi[buf][row * 64 + col], Ki + g);
      async_b128(&sVr[buf][row * 64 + col], Vr + g);
      async_b128(&sVi[buf][row * 64 + col], Vi + g);
    }
  };

  stageKV(0, 0);
  for (int kc = 0; kc < CS; kc += 32) {
    const int cur = (kc >> 5) & 1;
    wait_async0();        // my async copies for `cur` have landed
    __syncthreads();      // everyone's landed; prev reads of cur^1 done
    if (kc + 32 < CS) stageKV(cur ^ 1, kc + 32);   // prefetch next key chunk

    // Complex scores for 2 subtiles of 16 keys each.
    float mag[2][8];
#pragma unroll
    for (int st = 0; st < 2; ++st) {
      v8f sr = {}, sia = {}, sib = {};
#pragma unroll
      for (int c = 0; c < 2; ++c) {
        v16h bKr = frag_rm(&sKr[cur][(st * 16) * 64 + c * 32], 64);
        v16h bKi = frag_rm(&sKi[cur][(st * 16) * 64 + c * 32], 64);
        sr  = WMMA_F16(qfr[c], bKr, sr);   // qr . kr
        sr  = WMMA_F16(qfi[c], bKi, sr);   // + qi . ki
        sia = WMMA_F16(qfi[c], bKr, sia);  // qi . kr
        sib = WMMA_F16(qfr[c], bKi, sib);  // qr . ki (subtracted below)
      }
#pragma unroll
      for (int r = 0; r < 8; ++r) {
        float a  = sr[r];
        float bb = sia[r] - sib[r];
        mag[st][r] = ATT_SCALE * sqrtf(a * a + bb * bb);
      }
    }

    // Online softmax update (row-wise over the 16-lane group holding a row).
#pragma unroll
    for (int r = 0; r < 8; ++r) {
      float cm = fmaxf(mag[0][r], mag[1][r]);
#pragma unroll
      for (int msk = 1; msk < 16; msk <<= 1)
        cm = fmaxf(cm, __shfl_xor(cm, msk, 32));
      float nm   = fmaxf(rmax[r], cm);
      float corr = __expf(rmax[r] - nm);
      rmax[r] = nm;
      float p0 = __expf(mag[0][r] - nm);
      float p1 = __expf(mag[1][r] - nm);
      sP[w][(r + mb) * 32 + nl]      = (_Float16)p0;
      sP[w][(r + mb) * 32 + 16 + nl] = (_Float16)p1;
      float ps = p0 + p1;
#pragma unroll
      for (int msk = 1; msk < 16; msk <<= 1)
        ps += __shfl_xor(ps, msk, 32);
      rsum[r] = rsum[r] * corr + ps;
#pragma unroll
      for (int ni = 0; ni < 4; ++ni) { accOr[ni][r] *= corr; accOi[ni][r] *= corr; }
    }
    __syncthreads();   // P visible in LDS (also fences dscnt for this wave)

    // O += P @ V  (K-dim = 32 keys = one WMMA step; 4 subtiles of 16 dims).
    v16h pf = frag_rm(&sP[w][0], 32);
#pragma unroll
    for (int ni = 0; ni < 4; ++ni) {
      v16h bvr = frag_cm(&sVr[cur][ni * 16], 64);
      v16h bvi = frag_cm(&sVi[cur][ni * 16], 64);
      accOr[ni] = WMMA_F16(pf, bvr, accOr[ni]);
      accOi[ni] = WMMA_F16(pf, bvi, accOi[ni]);
    }
  }

  // Normalize and write merged-head output [b, m, h*64 + d] as f16.
#pragma unroll
  for (int ni = 0; ni < 4; ++ni)
#pragma unroll
    for (int r = 0; r < 8; ++r) {
      int m = q0 + w * 16 + r + mb;
      int d = ni * 16 + nl;
      size_t off = (size_t)(b * CS + m) * CE + hoff + d;
      float inv = 1.0f / rsum[r];
      Or[off] = (_Float16)(accOr[ni][r] * inv);
      Oi[off] = (_Float16)(accOi[ni][r] * inv);
    }
}

// ---------------------------------------------------------------------------
// Host launcher
// ---------------------------------------------------------------------------
extern "C" void kernel_launch(void* const* d_in, const int* in_sizes, int n_in,
                              void* d_out, int out_size, void* d_ws, size_t ws_size,
                              hipStream_t stream) {
  (void)in_sizes; (void)n_in; (void)out_size; (void)ws_size;

  const size_t ME = (size_t)CM * CE;   // 4096*512
  const size_t EE = (size_t)CE * CE;   // 512*512

  _Float16* ws = (_Float16*)d_ws;
  _Float16* xr16 = ws;  ws += ME;
  _Float16* xi16 = ws;  ws += ME;
  _Float16* w16[8];
  for (int i = 0; i < 8; ++i) { w16[i] = ws; ws += EE; }
  _Float16* qr16 = ws; ws += ME;
  _Float16* qi16 = ws; ws += ME;
  _Float16* kr16 = ws; ws += ME;
  _Float16* ki16 = ws; ws += ME;
  _Float16* vr16 = ws; ws += ME;
  _Float16* vi16 = ws; ws += ME;
  _Float16* ar16 = ws; ws += ME;
  _Float16* ai16 = ws; ws += ME;

  // 1) fp32 -> f16 conversions
  cvt_f32_to_f16<<<(int)(ME / 256), 256, 0, stream>>>((const float*)d_in[0], xr16, (int)ME);
  cvt_f32_to_f16<<<(int)(ME / 256), 256, 0, stream>>>((const float*)d_in[1], xi16, (int)ME);
  const int widx[8] = {2, 3, 6, 7, 10, 11, 14, 15};  // q_wr,q_wi,k_wr,k_wi,v_wr,v_wi,o_wr,o_wi
  for (int i = 0; i < 8; ++i)
    cvt_f32_to_f16<<<(int)(EE / 256), 256, 0, stream>>>((const float*)d_in[widx[i]], w16[i], (int)EE);

  // 2) Q/K/V complex projections (f16 outputs)
  dim3 ggrid(CM / 128, CE / 64);
  cplx_gemm_kernel<<<ggrid, 256, 0, stream>>>(xr16, xi16, w16[0], w16[1],
      (const float*)d_in[4], (const float*)d_in[5], qr16, qi16, nullptr, nullptr);
  cplx_gemm_kernel<<<ggrid, 256, 0, stream>>>(xr16, xi16, w16[2], w16[3],
      (const float*)d_in[8], (const float*)d_in[9], kr16, ki16, nullptr, nullptr);
  cplx_gemm_kernel<<<ggrid, 256, 0, stream>>>(xr16, xi16, w16[4], w16[5],
      (const float*)d_in[12], (const float*)d_in[13], vr16, vi16, nullptr, nullptr);

  // 3) Complex-magnitude flash attention
  dim3 agrid(CB * CH, CS / 64);
  cplx_attn_kernel<<<agrid, 128, 0, stream>>>(qr16, qi16, kr16, ki16, vr16, vi16, ar16, ai16);

  // 4) Output complex projection, fp32 straight into d_out (yr then yi)
  float* yr = (float*)d_out;
  float* yi = yr + ME;
  cplx_gemm_kernel<<<ggrid, 256, 0, stream>>>(ar16, ai16, w16[6], w16[7],
      (const float*)d_in[16], (const float*)d_in[17], nullptr, nullptr, yr, yi);
}
